// RGNN_55611236549297
// MI455X (gfx1250) — matrix operand order repaired and checked by reading, not compiled, not run on hardware
//
#include <hip/hip_runtime.h>
#include <hip/hip_bf16.h>

typedef __attribute__((ext_vector_type(16))) _Float16 v16h;
typedef __attribute__((ext_vector_type(8)))  float    v8f;

#define TEAMS_PER_BLOCK 128   // 8 waves x 16 rows
#define SWZ_GATE  8192        // halves per swizzled gate weight (128x64)
#define SWZ_W0    1024        // halves for swizzled W0cat (64x16)

// ---------------------------------------------------------------------------
// Prep: f16-convert + B-fragment-swizzle the gate weights and W0cat, fold biases.
// Swizzled layout: frag index f = ((nt*4+kk)*2+grp)*16 + (lane&15), 16 halves
// contiguous per (frag,lane) so a wave loads its whole B fragment as v16h.
// B k order used here (contiguous, same as A; 32K x 16N): element j -> K = kk*32 + grp*16 + j,
// N = nt*16 + (lane&15).
// ---------------------------------------------------------------------------
__global__ __launch_bounds__(256) void prep_kernel(
    const float* __restrict__ w_xz, const float* __restrict__ w_xh,
    const float* __restrict__ W0,
    const float* __restrict__ b_xz, const float* __restrict__ b_hz,
    const float* __restrict__ b_xh, const float* __restrict__ b_hh,
    _Float16* __restrict__ wzSw, _Float16* __restrict__ whSw,
    _Float16* __restrict__ w0Sw, float* __restrict__ bz, float* __restrict__ bh)
{
    int idx = blockIdx.x * 256 + threadIdx.x;
    if (idx < SWZ_GATE) {
        int j   = idx & 15;
        int n   = (idx >> 4) & 15;
        int grp = (idx >> 8) & 1;
        int kk  = (idx >> 9) & 3;
        int nt  = idx >> 11;
        int K = kk * 32 + grp * 16 + j;      // 0..127
        int N = nt * 16 + n;                 // 0..63
        wzSw[idx] = (_Float16)w_xz[K * 64 + N];
        whSw[idx] = (_Float16)w_xh[K * 64 + N];
    } else if (idx < SWZ_GATE + SWZ_W0) {
        int d   = idx - SWZ_GATE;
        int j   = d & 15;
        int n   = (d >> 4) & 15;
        int grp = (d >> 8) & 1;
        int kk  = d >> 9;                    // 0..1
        int K = kk * 32 + grp * 16 + j;      // 0..63
        // W0cat: col n<8 -> W0[K][n] (top half), n>=8 -> W0[64+K][n-8] (bottom)
        float v = (n < 8) ? W0[K * 8 + n] : W0[(64 + K) * 8 + (n - 8)];
        w0Sw[d] = (_Float16)v;
    } else if (idx < SWZ_GATE + SWZ_W0 + 64) {
        int i = idx - (SWZ_GATE + SWZ_W0);
        bz[i] = b_xz[i] + b_hz[i];
    } else if (idx < SWZ_GATE + SWZ_W0 + 128) {
        int i = idx - (SWZ_GATE + SWZ_W0 + 64);
        bh[i] = b_xh[i] + b_hh[i];
    }
}

// ---------------------------------------------------------------------------
// Node phase: per wave, 16 nodes. Two f16 WMMA GEMM chains (z-gate, h-tilde)
// over K=128 (4 steps) x N=64 (4 tiles), gate math, then a fused 16x64x16
// WMMA projection p = h @ [W0_top | W0_bot] (the edge MLP's first layer,
// factored per node). Output hP[t][0..7]=p_top, [8..15]=p_bot.
// ---------------------------------------------------------------------------
__global__ __launch_bounds__(256) void node_kernel(
    const float* __restrict__ emb,
    const _Float16* __restrict__ wzSw, const _Float16* __restrict__ whSw,
    const _Float16* __restrict__ w0Sw,
    const float* __restrict__ bz, const float* __restrict__ bh,
    float* __restrict__ hP, int T)
{
    __shared__ float ldsH[8][16][65];  // per-wave h tile, padded stride

    const int lane = threadIdx.x & 31;
    const int wave = threadIdx.x >> 5;
    const int grp  = lane >> 4;       // half-wave group (0|1)
    const int l15  = lane & 15;
    const int R    = blockIdx.x * TEAMS_PER_BLOCK + wave * 16;

    int arowIdx = R + l15;            // A layout: M = lane&15
    if (arowIdx >= T) arowIdx = T - 1;                 // clamp; stores guarded
    const float* __restrict__ arow = emb + (size_t)arowIdx * 128;

    v8f accZ[4] = {};
    v8f accH[4] = {};

    #pragma unroll
    for (int kk = 0; kk < 4; ++kk) {
        // A fragment: contiguous k order, same as B swizzle. VGPR v: K=2v,2v+1 ;
        // upper half-wave adds +16. All K0 even -> float2 ok.
        v16h a;
        #pragma unroll
        for (int v = 0; v < 8; ++v) {
            int K0 = 2 * v + grp * 16 + kk * 32;
            float2 f = *(const float2*)(arow + K0);
            a[2 * v]     = (_Float16)f.x;
            a[2 * v + 1] = (_Float16)f.y;
        }
        #pragma unroll
        for (int nt = 0; nt < 4; ++nt) {
            int fbase = ((((nt * 4 + kk) * 2 + grp) * 16) + l15) * 16;
            v16h bzf = *(const v16h*)(wzSw + fbase);   // 32B -> 2x b128
            accZ[nt] = __builtin_amdgcn_wmma_f32_16x16x32_f16(
                false, a, false, bzf, (short)0, accZ[nt], false, false);
            v16h bhf = *(const v16h*)(whSw + fbase);
            accH[nt] = __builtin_amdgcn_wmma_f32_16x16x32_f16(
                false, a, false, bhf, (short)0, accH[nt], false, false);
        }
    }

    // Gate math + stage h (D layout: M = r + 8*grp, N = nt*16 + l15) into LDS.
    #pragma unroll
    for (int nt = 0; nt < 4; ++nt) {
        int N = nt * 16 + l15;
        float bzv = bz[N];
        float bhv = bh[N];
        #pragma unroll
        for (int r = 0; r < 8; ++r) {
            int m = r + grp * 8;
            float z  = 1.0f / (1.0f + __expf(-(accZ[nt][r] + bzv)));
            float ht = tanhf(accH[nt][r] + bhv);
            float h  = (1.0f - z) * ht;
            ldsH[wave][m][N] = (h > 0.0f) ? h : 0.0f;   // relu
        }
    }
    __syncthreads();

    // p = h @ W0cat : 16x64x16 via two K=32 WMMA steps.
    v8f pAcc = {};
    #pragma unroll
    for (int kk = 0; kk < 2; ++kk) {
        v16h a;
        #pragma unroll
        for (int v = 0; v < 8; ++v) {
            int K0 = 2 * v + grp * 16 + kk * 32;
            a[2 * v]     = (_Float16)ldsH[wave][l15][K0];
            a[2 * v + 1] = (_Float16)ldsH[wave][l15][K0 + 1];
        }
        v16h bf = *(const v16h*)(w0Sw + (((kk * 2 + grp) * 16) + l15) * 16);
        pAcc = __builtin_amdgcn_wmma_f32_16x16x32_f16(
            false, a, false, bf, (short)0, pAcc, false, false);
    }

    #pragma unroll
    for (int r = 0; r < 8; ++r) {
        int t = R + r + grp * 8;
        if (t < T) hP[(size_t)t * 16 + l15] = pAcc[r];
    }
}

// ---------------------------------------------------------------------------
// Edge phase: gather precomputed 8-float projections (32B per endpoint,
// float4-vectorized), tiny 8-wide relu MLP from LDS-broadcast weights, softmax.
// ---------------------------------------------------------------------------
__global__ __launch_bounds__(256) void edge_kernel(
    const int* __restrict__ ei, const float* __restrict__ hP,
    const float* __restrict__ W1, const float* __restrict__ W2,
    const float* __restrict__ W3, const float* __restrict__ W4,
    const float* __restrict__ W5,
    const float* __restrict__ b0, const float* __restrict__ b1,
    const float* __restrict__ b2, const float* __restrict__ b3,
    const float* __restrict__ b4, const float* __restrict__ b5,
    float* __restrict__ out, int E)
{
    __shared__ float sW[4][64];
    __shared__ float sW5[24];
    __shared__ float sB[5][8];
    __shared__ float sB5[3];

    const int t = threadIdx.x;
    if (t < 64) { sW[0][t] = W1[t]; sW[1][t] = W2[t]; sW[2][t] = W3[t]; sW[3][t] = W4[t]; }
    if (t < 24) sW5[t] = W5[t];
    if (t < 8)  { sB[0][t] = b0[t]; sB[1][t] = b1[t]; sB[2][t] = b2[t];
                  sB[3][t] = b3[t]; sB[4][t] = b4[t]; }
    if (t < 3)  sB5[t] = b5[t];
    __syncthreads();

    const int e = blockIdx.x * 256 + t;
    if (e >= E) return;

    const int home = ei[e];
    const int away = ei[E + e];
    const float4* ph = (const float4*)(hP + (size_t)home * 16);      // p_top
    const float4* pa = (const float4*)(hP + (size_t)away * 16 + 8);  // p_bot
    float4 h0 = ph[0], h1 = ph[1];
    float4 a0 = pa[0], a1 = pa[1];

    float f[8], g[8];
    f[0] = h0.x + a0.x; f[1] = h0.y + a0.y; f[2] = h0.z + a0.z; f[3] = h0.w + a0.w;
    f[4] = h1.x + a1.x; f[5] = h1.y + a1.y; f[6] = h1.z + a1.z; f[7] = h1.w + a1.w;
    #pragma unroll
    for (int j = 0; j < 8; ++j) {
        float v = f[j] + sB[0][j];
        f[j] = (v > 0.0f) ? v : 0.0f;                 // layer 0 (refactored)
    }

    #pragma unroll
    for (int l = 0; l < 4; ++l) {                      // layers 1..4 (8x8)
        #pragma unroll
        for (int j = 0; j < 8; ++j) {
            float acc = sB[l + 1][j];
            #pragma unroll
            for (int k = 0; k < 8; ++k) acc = fmaf(f[k], sW[l][k * 8 + j], acc);
            g[j] = (acc > 0.0f) ? acc : 0.0f;
        }
        #pragma unroll
        for (int j = 0; j < 8; ++j) f[j] = g[j];
    }

    float o[3];
    #pragma unroll
    for (int c = 0; c < 3; ++c) {                      // layer 5 (8->3) + relu
        float acc = sB5[c];
        #pragma unroll
        for (int k = 0; k < 8; ++k) acc = fmaf(f[k], sW5[k * 3 + c], acc);
        o[c] = (acc > 0.0f) ? acc : 0.0f;
    }
    float m  = fmaxf(o[0], fmaxf(o[1], o[2]));
    float e0 = __expf(o[0] - m), e1 = __expf(o[1] - m), e2 = __expf(o[2] - m);
    float inv = 1.0f / (e0 + e1 + e2);
    out[(size_t)e * 3 + 0] = e0 * inv;
    out[(size_t)e * 3 + 1] = e1 * inv;
    out[(size_t)e * 3 + 2] = e2 * inv;
}

// ---------------------------------------------------------------------------
// Input order (setup_inputs dict order):
//  0 edge_index[2,E] i32 | 1 embedding[T,128] f32
//  2 w_xz 3 b_xz 4 w_xr 5 b_xr 6 w_xh 7 b_xh
//  8 w_hz 9 b_hz 10 w_hr 11 b_hr 12 w_hh 13 b_hh
//  14..19 dense_ws (W0[128x8], W1..W4[8x8], W5[8x3])
//  20..25 dense_bs (b0..b4[8], b5[3])
// ---------------------------------------------------------------------------
extern "C" void kernel_launch(void* const* d_in, const int* in_sizes, int n_in,
                              void* d_out, int out_size, void* d_ws, size_t ws_size,
                              hipStream_t stream) {
    (void)n_in; (void)out_size; (void)ws_size;

    const int E = in_sizes[0] / 2;
    const int T = in_sizes[1] / 128;

    const int*   ei   = (const int*)d_in[0];
    const float* emb  = (const float*)d_in[1];
    const float* w_xz = (const float*)d_in[2];
    const float* b_xz = (const float*)d_in[3];
    const float* w_xh = (const float*)d_in[6];
    const float* b_xh = (const float*)d_in[7];
    const float* b_hz = (const float*)d_in[9];
    const float* b_hh = (const float*)d_in[13];
    const float* W0   = (const float*)d_in[14];
    const float* W1   = (const float*)d_in[15];
    const float* W2   = (const float*)d_in[16];
    const float* W3   = (const float*)d_in[17];
    const float* W4   = (const float*)d_in[18];
    const float* W5   = (const float*)d_in[19];
    const float* b0   = (const float*)d_in[20];
    const float* b1   = (const float*)d_in[21];
    const float* b2   = (const float*)d_in[22];
    const float* b3   = (const float*)d_in[23];
    const float* b4   = (const float*)d_in[24];
    const float* b5   = (const float*)d_in[25];
    float* out = (float*)d_out;

    // Workspace carving (base is 256B-aligned; offsets keep 32B alignment).
    float*    hP   = (float*)d_ws;                       // T*16 f32
    _Float16* wzSw = (_Float16*)(hP + (size_t)T * 16);   // 8192 halves
    _Float16* whSw = wzSw + SWZ_GATE;                    // 8192 halves
    _Float16* w0Sw = whSw + SWZ_GATE;                    // 1024 halves
    float*    bz   = (float*)(w0Sw + SWZ_W0);            // 64 f32
    float*    bh   = bz + 64;                            // 64 f32

    {
        int work = SWZ_GATE + SWZ_W0 + 128;
        int blocks = (work + 255) / 256;
        prep_kernel<<<blocks, 256, 0, stream>>>(w_xz, w_xh, W0, b_xz, b_hz,
                                                b_xh, b_hh, wzSw, whSw, w0Sw,
                                                bz, bh);
    }
    {
        int blocks = (T + TEAMS_PER_BLOCK - 1) / TEAMS_PER_BLOCK;
        node_kernel<<<blocks, 256, 0, stream>>>(emb, wzSw, whSw, w0Sw, bz, bh,
                                                hP, T);
    }
    {
        int blocks = (E + 255) / 256;
        edge_kernel<<<blocks, 256, 0, stream>>>(ei, hP, W1, W2, W3, W4, W5,
                                                b0, b1, b2, b3, b4, b5, out, E);
    }
}
